// SelfAttention_39926015983833
// MI455X (gfx1250) — matrix operand order repaired and checked by reading, not compiled, not run on hardware
//
#include <hip/hip_runtime.h>
#include <hip/hip_bf16.h>

// ---------------------------------------------------------------------------
// Fused pooled self-attention for MI455X (gfx1250, wave32, WMMA).
// B=16, C=128, H=W=64 -> N=4096 queries, M=1024 pooled keys per batch.
// Flash-style fusion: the [N x M] attention matrix (268 MB fp32) never touches
// HBM; only x in (33.5MB) + out (33.5MB) + ~4.5MB f16 staging do. Roofline:
// ~2.9us of HBM at 23.3 TB/s vs ~12.4 GFLOP on f16 WMMA -> balanced, so the
// loop stays WMMA-dense: softmax row sums are computed BY WMMA (ones-column
// trick) and the tile-global max reduction runs on DPP/permlane (no LDS).
// ---------------------------------------------------------------------------

#define Bc 16
#define Cc 128
#define Hc 64
#define Wc 64
#define Nc 4096   // H*W
#define Mc 1024   // (H/2)*(W/2)
#define CQ 16     // C/8
#define CV 64     // C/2

typedef _Float16 v2h  __attribute__((ext_vector_type(2)));
typedef _Float16 v8h  __attribute__((ext_vector_type(8)));
typedef _Float16 v16h __attribute__((ext_vector_type(16)));
typedef float    v8f  __attribute__((ext_vector_type(8)));

__device__ __forceinline__ v8h ld8(const _Float16* p) { return *(const v8h*)p; }

__device__ __forceinline__ v16h cat16(v8h lo, v8h hi) {
  v16h r;
#pragma unroll
  for (int i = 0; i < 8; ++i) { r[i] = lo[i]; r[i + 8] = hi[i]; }
  return r;
}
__device__ __forceinline__ v16h ld16(const _Float16* p) { return cat16(ld8(p), ld8(p + 8)); }
__device__ __forceinline__ v8f zero8f() {
  v8f r;
#pragma unroll
  for (int i = 0; i < 8; ++i) r[i] = 0.f;
  return r;
}
__device__ __forceinline__ v16h ones16h() {
  v16h r;
#pragma unroll
  for (int i = 0; i < 16; ++i) r[i] = (_Float16)1.f;
  return r;
}

__device__ __forceinline__ v8f wmma_f16(v16h a, v16h b, v8f c) {
  // v_wmma_f32_16x16x32_f16: D = A(16x32 f16) * B(32x16 f16) + C(16x16 f32)
  return __builtin_amdgcn_wmma_f32_16x16x32_f16(false, a, false, b, (short)0, c, false, false);
}

__device__ __forceinline__ void lds_fence() {
  asm volatile("s_wait_dscnt 0" ::: "memory");
}

// Pack two f32 -> packed f16 pair (v_cvt_pk_rtz_f16_f32), bit-cast to v2h.
__device__ __forceinline__ v2h pack_f16(float a, float b) {
  return __builtin_bit_cast(v2h, __builtin_amdgcn_cvt_pkrtz(a, b));
}

// fmax with the lane xor'd by DPP ROW_XMASK (VALU, no LDS round trip).
// DPP control must be a literal constant -> template parameter.
template <int CTL>
__device__ __forceinline__ float max_xor_dpp(float v) {
  int s = __builtin_bit_cast(int, v);
  int t = __builtin_amdgcn_update_dpp(s, s, CTL, 0xf, 0xf, true);
  return fmaxf(v, __builtin_bit_cast(float, t));
}
// Full wave32 max: xor 1,2,4,8 via ROW_XMASK, xor 16 via v_permlanex16.
__device__ __forceinline__ float wave_max32(float v) {
  v = max_xor_dpp<0x161>(v);
  v = max_xor_dpp<0x162>(v);
  v = max_xor_dpp<0x164>(v);
  v = max_xor_dpp<0x168>(v);
  int s = __builtin_bit_cast(int, v);
  int t = __builtin_amdgcn_permlanex16(s, s, 0x76543210, 0xfedcba98, false, false);
  return fmaxf(v, __builtin_bit_cast(float, t));
}

// ---------------------------------------------------------------------------
// Prep 1: Q projection.  q16[b][n][cq] (f16), one thread per (b,n).
// ---------------------------------------------------------------------------
__global__ __launch_bounds__(256) void qproj_kernel(
    const float* __restrict__ x, const float* __restrict__ wq,
    const float* __restrict__ bq, _Float16* __restrict__ q16) {
  int gid = blockIdx.x * 256 + threadIdx.x;        // Bc*Nc threads
  int b = gid >> 12, n = gid & (Nc - 1);
  const float* xb = x + ((size_t)b * Cc) * Nc + n;
  float acc[CQ];
#pragma unroll
  for (int i = 0; i < CQ; ++i) acc[i] = bq[i];
  for (int c = 0; c < Cc; ++c) {
    float xv = xb[(size_t)c * Nc];                 // coalesced along n
#pragma unroll
    for (int i = 0; i < CQ; ++i) acc[i] = fmaf(wq[i * Cc + c], xv, acc[i]); // SGPR weights
  }
  _Float16* qp = q16 + (size_t)gid * CQ;
#pragma unroll
  for (int i = 0; i < CQ; ++i) qp[i] = (_Float16)acc[i];
}

// ---------------------------------------------------------------------------
// Prep 2: fused 2x2 maxpool + K and V projections.
// k16[b][m][cq] (contiguous cq -> WMMA-B loads), v16[b][cv][m] (contiguous m).
// ---------------------------------------------------------------------------
__global__ __launch_bounds__(256) void kvproj_kernel(
    const float* __restrict__ x, const float* __restrict__ wk,
    const float* __restrict__ bk, const float* __restrict__ wv,
    const float* __restrict__ bv, _Float16* __restrict__ k16,
    _Float16* __restrict__ v16) {
  int gid = blockIdx.x * 256 + threadIdx.x;        // Bc*Mc threads
  int b = gid >> 10, m = gid & (Mc - 1);
  int ph = m >> 5, pw = m & 31;
  const float* xb = x + ((size_t)b * Cc) * Nc + ph * 2 * Wc + pw * 2;
  float ka[CQ], va[CV];
#pragma unroll
  for (int i = 0; i < CQ; ++i) ka[i] = bk[i];
#pragma unroll
  for (int i = 0; i < CV; ++i) va[i] = bv[i];
  for (int c = 0; c < Cc; ++c) {
    const float* xc = xb + (size_t)c * Nc;
    float p = fmaxf(fmaxf(xc[0], xc[1]), fmaxf(xc[Wc], xc[Wc + 1]));
#pragma unroll
    for (int i = 0; i < CQ; ++i) ka[i] = fmaf(wk[i * Cc + c], p, ka[i]);
#pragma unroll 16
    for (int i = 0; i < CV; ++i) va[i] = fmaf(wv[i * Cc + c], p, va[i]);
  }
  _Float16* kp = k16 + (size_t)gid * CQ;
#pragma unroll
  for (int i = 0; i < CQ; ++i) kp[i] = (_Float16)ka[i];
#pragma unroll 16
  for (int i = 0; i < CV; ++i) v16[((size_t)b * CV + i) * Mc + m] = (_Float16)va[i];
}

// ---------------------------------------------------------------------------
// Prep 3: wa -> f16 copy (layout [co][cv]) + zero-pad page for grp1 K loads.
// ---------------------------------------------------------------------------
__global__ __launch_bounds__(256) void wacvt_kernel(const float* __restrict__ wa,
                                                    _Float16* __restrict__ wa16,
                                                    _Float16* __restrict__ zpad) {
  int i = blockIdx.x * 256 + threadIdx.x;
  if (i < Cc * CV) wa16[i] = (_Float16)wa[i];
  if (i < 32) zpad[i] = (_Float16)0.f;
}

// ---------------------------------------------------------------------------
// Main: flash attention + fused output conv + residual.
// One wave = one 16-query row block. Per 32-key iteration: 2 logits WMMAs,
// tile-global online max (DPP, uniform-branch rescale skip), exp, packed P
// transpose via LDS (8 b32 stores / 2 b128 loads per lane), 4 feat WMMAs +
// 1 ones-column WMMA accumulating the softmax row sums in D-layout.
// Epilogue: 1/l scale, F->LDS, 16 WMMAs for the 64->128 conv (wa as A ->
// coalesced n stores), +bias, *gamma, +x residual.
// ---------------------------------------------------------------------------
__global__ __launch_bounds__(256) void attn_kernel(
    const float* __restrict__ x, const _Float16* __restrict__ q16,
    const _Float16* __restrict__ k16, const _Float16* __restrict__ v16,
    const _Float16* __restrict__ wa16, const _Float16* __restrict__ zpad,
    const float* __restrict__ ba, const float* __restrict__ gamma,
    float* __restrict__ out) {
  __shared__ _Float16 stage[8 * 16 * 72];          // per-wave 16 rows x 72 halves

  const int lane = threadIdx.x & 31;
  const int wave = threadIdx.x >> 5;
  const int col  = lane & 15;                      // N index (B/D) / row index (A)
  const int grp  = lane >> 4;                      // half-wave (K split)
  const int rb   = blockIdx.x * 8 + wave;          // global 16-query row block
  const int b    = rb >> 8;
  const int n0   = (rb & 255) << 4;
  _Float16* st = stage + wave * (16 * 72);

  // A(Q): 16x32 f16, lane=query row, K=channel (16 real + 16 zero pad).
  // A layout: grp0 halves 0..15 -> K0..15; grp1 halves 0..15 -> K16..31 (zero pad).
  v16h aq;
  {
    const _Float16* qp = (grp == 0) ? q16 + ((size_t)b * Nc + n0 + col) * CQ : zpad;
    aq = ld16(qp);
  }

  // Branchless K pointers: grp1 lanes read the zero page with zero stride
  // (B upper-half K pad), so the loop body has no exec-mask churn.
  const _Float16* kptr0 = (grp == 0) ? k16 + ((size_t)b * Mc + col) * CQ      : zpad;
  const _Float16* kptr1 = (grp == 0) ? k16 + ((size_t)b * Mc + 16 + col) * CQ : zpad;
  const size_t    kstep = (grp == 0) ? (size_t)32 * CQ : 0;

  const v8f  z8   = zero8f();
  const v16h oneB = ones16h();                     // all-ones B: row-sum generator
  v8f facc[4];
#pragma unroll
  for (int j = 0; j < 4; ++j) facc[j] = zero8f();
  v8f   lacc = zero8f();                           // softmax denominators (D-layout)
  float mrun = -1.0e30f;                           // running tile-global max

  for (int t = 0; t < Mc / 32; ++t) {              // 32 iterations, 32 keys each
    const int m0 = t * 32;
    __builtin_prefetch(kptr0 + kstep, 0, 1);       // global_prefetch_b8 next tiles
    __builtin_prefetch(v16 + ((size_t)b * CV + col) * Mc + m0 + 32, 0, 1);

    v16h kb0 = ld16(kptr0);                        // B(K): lane&15=key, halves=chan
    v16h kb1 = ld16(kptr1);
    kptr0 += kstep; kptr1 += kstep;
    v8f s0 = wmma_f16(aq, kb0, z8);                // S[query][key] tiles
    v8f s1 = wmma_f16(aq, kb1, z8);

    // Tile-global max (softmax is invariant to any per-row constant; one shared
    // constant is valid and values span only a few units -> no underflow).
    float tm = fmaxf(s0[0], s1[0]);
#pragma unroll
    for (int r = 1; r < 8; ++r) tm = fmaxf(tm, fmaxf(s0[r], s1[r]));
    tm = wave_max32(tm);                           // DPP + permlanex16, no LDS
    const float nm = fmaxf(mrun, tm);
    if (nm > mrun) {                               // wave-uniform branch: rescale
      const float sc = __expf(mrun - nm);          // first iter: exp(-inf) = 0
#pragma unroll
      for (int j = 0; j < 4; ++j)
#pragma unroll
        for (int r = 0; r < 8; ++r) facc[j][r] *= sc;
#pragma unroll
      for (int r = 0; r < 8; ++r) lacc[r] *= sc;
      mrun = nm;
    }

    float p0[8], p1[8];
#pragma unroll
    for (int r = 0; r < 8; ++r) { p0[r] = __expf(s0[r] - nm); p1[r] = __expf(s1[r] - nm); }

    // Transpose P (D layout) -> A layout via per-wave LDS. Word layout:
    // word[row][c] = pack_f16(P[row][key=c], P[row][key=c+16]); row stride
    // 20 words (80B) keeps b128 reloads 16B-aligned and banks spread.
    lds_fence();
#pragma unroll
    for (int r = 0; r < 8; ++r) {
      int row = r + 8 * grp;
      *(v2h*)(st + row * 40 + col * 2) = pack_f16(p0[r], p1[r]);
    }
    lds_fence();
    // A reload: lane = query row; grp selects words 0..7 / 8..15.
    // lo halves -> keys c (+grp*8), hi halves -> keys c+16.
    const _Float16* rp = st + col * 40 + grp * 16;
    v8h w0 = ld8(rp), w1 = ld8(rp + 8);
    v16h ap;
#pragma unroll
    for (int i = 0; i < 4; ++i) {
      ap[i]      = w0[2 * i]; ap[8 + i]  = w0[2 * i + 1];
      ap[4 + i]  = w1[2 * i]; ap[12 + i] = w1[2 * i + 1];
    }

#pragma unroll
    for (int j = 0; j < 4; ++j) {                  // B(V): lane&15 = cv, halves = keys
      const _Float16* vp = v16 + ((size_t)b * CV + j * 16 + col) * Mc + m0 + grp * 8;
      facc[j] = wmma_f16(ap, cat16(ld8(vp), ld8(vp + 16)), facc[j]);
    }
    lacc = wmma_f16(ap, oneB, lacc);               // row sums by matrix op
  }

  // Normalize by row sums (lacc holds l for row r+8*grp, replicated per group),
  // stage F[query][cv] in LDS as f16.
  float rl[8];
#pragma unroll
  for (int r = 0; r < 8; ++r) rl[r] = 1.0f / lacc[r];
  lds_fence();
#pragma unroll
  for (int j = 0; j < 4; ++j)
#pragma unroll
    for (int r = 0; r < 8; ++r) {
      int row = r + 8 * grp;
      st[row * 72 + j * 16 + col] = (_Float16)(facc[j][r] * rl[r]);
    }
  lds_fence();

  // Final conv as out^T tile: A = wa (16 co x 32 cv), B = F (32 cv x 16 queries)
  // -> D rows are fixed co, lanes sweep n -> coalesced 64B loads/stores.
  const _Float16* fp = st + col * 72;              // lane = query column for B
  v16h bf0 = ld16(fp + grp * 16);                  // cv 0..31
  v16h bf1 = ld16(fp + 32 + grp * 16);             // cv 32..63
  const float g = gamma[0];

#pragma unroll
  for (int j2 = 0; j2 < 8; ++j2) {                 // 128 output channels / 16
    const _Float16* wp = wa16 + (size_t)(j2 * 16 + col) * CV;   // lane = co row for A
    v16h aw0 = ld16(wp + grp * 16);
    v16h aw1 = ld16(wp + 32 + grp * 16);
    v8f o = wmma_f16(aw0, bf0, z8);
    o = wmma_f16(aw1, bf1, o);
#pragma unroll
    for (int r = 0; r < 8; ++r) {
      int co = j2 * 16 + r + 8 * grp;
      size_t idx = ((size_t)b * Cc + co) * Nc + n0 + col;
      out[idx] = g * (o[r] + ba[co]) + x[idx];
    }
  }
}

// ---------------------------------------------------------------------------
extern "C" void kernel_launch(void* const* d_in, const int* in_sizes, int n_in,
                              void* d_out, int out_size, void* d_ws, size_t ws_size,
                              hipStream_t stream) {
  const float* x     = (const float*)d_in[0];
  const float* wq    = (const float*)d_in[1];
  const float* bq    = (const float*)d_in[2];
  const float* wk    = (const float*)d_in[3];
  const float* bk    = (const float*)d_in[4];
  const float* wv    = (const float*)d_in[5];
  const float* bv    = (const float*)d_in[6];
  const float* wa    = (const float*)d_in[7];
  const float* ba    = (const float*)d_in[8];
  const float* gamma = (const float*)d_in[9];
  float* out = (float*)d_out;

  char* ws = (char*)d_ws;                          // ~4.52 MB of f16 staging
  _Float16* q16  = (_Float16*)(ws);                                  // 2 MB
  _Float16* k16  = (_Float16*)(ws + (2u << 20));                     // 512 KB
  _Float16* v16  = (_Float16*)(ws + (2u << 20) + (512u << 10));      // 2 MB
  _Float16* wa16 = (_Float16*)(ws + (4u << 20) + (512u << 10));      // 16 KB
  _Float16* zpad = (_Float16*)(ws + (4u << 20) + (528u << 10));      // 64 B zeros

  qproj_kernel<<<(Bc * Nc) / 256, 256, 0, stream>>>(x, wq, bq, q16);
  kvproj_kernel<<<(Bc * Mc) / 256, 256, 0, stream>>>(x, wk, bk, wv, bv, k16, v16);
  wacvt_kernel<<<(Cc * CV + 255) / 256, 256, 0, stream>>>(wa, wa16, zpad);
  // 4096 row blocks (16 queries each), 8 waves per 256-thread block.
  attn_kernel<<<(Bc * Nc / 16) / 8, 256, 0, stream>>>(x, q16, k16, v16, wa16, zpad,
                                                      ba, gamma, out);
}
